// GraphConvolution_7876970021469
// MI455X (gfx1250) — matrix operand order repaired and checked by reading, not compiled, not run on hardware
//
#include <hip/hip_runtime.h>
#include <hip/hip_bf16.h>
#include <stdint.h>

typedef __attribute__((ext_vector_type(16))) __bf16 v16bf;
typedef __attribute__((ext_vector_type(8)))  float  v8f;

#define FIN  512
#define FOUT 128
// wfrag: (FOUT/16 tiles) x (FIN/32 ksteps) x 32 lanes x 16 bf16 = 128 KB
#define WFRAG_HALFS ((FOUT / 16) * (FIN / 32) * 32 * 16)
#define WFRAG_BYTES (WFRAG_HALFS * 2)

// Native f32 -> bf16 (fptrunc, round-to-nearest-even) — lowers to the
// hardware bf16 convert ops on gfx1250 instead of manual bit-twiddling.
__device__ __forceinline__ __bf16 f2bf(float x) { return (__bf16)x; }

// ---------------------------------------------------------------------------
// Pre-swizzle W (FINxFOUT f32) into WMMA B-fragment layout, bf16.
// frag f = tile_n*16 + kstep; per frag: 32 lanes x 16 halfs.
// B 32x16 layout: lane l -> N = l&15 ; K = kstep*32 + (l<16?0:8) + (h<8 ? h : h+8)
// ---------------------------------------------------------------------------
__global__ void __launch_bounds__(256)
wprep_kernel(const float* __restrict__ W, uint16_t* __restrict__ wfrag) {
  int t    = blockIdx.x * blockDim.x + threadIdx.x;
  int lane = t & 31;
  int f    = t >> 5;
  if (f >= (FOUT / 16) * (FIN / 32)) return;
  int tile_n = f >> 4;                    // 0..7
  int kstep  = f & 15;                    // 0..15
  int col    = tile_n * 16 + (lane & 15);
  int kbase  = kstep * 32 + ((lane < 16) ? 0 : 8);
  uint16_t* dst = wfrag + (((size_t)f * 32) + lane) * 16;
#pragma unroll
  for (int h = 0; h < 16; ++h) {
    union { __bf16 b; uint16_t s; } u;
    u.b = f2bf(W[(size_t)(kbase + (h < 8 ? h : h + 8)) * FOUT + col]);
    dst[h] = u.s;
  }
}

// ---------------------------------------------------------------------------
// pre_sup = x @ W  via v_wmma_f32_16x16x32_bf16.
// Workgroup = 8 waves; wave w owns rows [blk*256 + w*32, +32) x all 128 cols.
// B fragments staged once per workgroup in LDS (128 KB of the WGP's 320 KB).
// Per k-step: batch-load all 8 B fragments (16x ds_load_b128) into registers,
// then fire 16 WMMAs back-to-back while the next A-stream clause is in flight.
// ---------------------------------------------------------------------------
__global__ void __launch_bounds__(256)
gemm_kernel(const float* __restrict__ x, const uint16_t* __restrict__ wfrag,
            float* __restrict__ pre, int n_rows) {
  extern __shared__ uint8_t lds_raw[];    // WFRAG_BYTES dynamic LDS

  // cooperative global -> LDS stage of the swizzled W fragments
  {
    const uint4* src = (const uint4*)wfrag;
    uint4*       dst = (uint4*)lds_raw;
    const int    n16 = WFRAG_BYTES / 16;  // 8192 x 16B
#pragma unroll 4
    for (int i = threadIdx.x; i < n16; i += 256) dst[i] = src[i];
  }
  __syncthreads();

  const int lane  = threadIdx.x & 31;
  const int wave  = threadIdx.x >> 5;
  const int mbase = blockIdx.x * 256 + wave * 32;
  const int half  = lane >> 4;            // 0: lanes 0-15, 1: lanes 16-31
  const int l15   = lane & 15;

  int arow0 = mbase + l15;
  int arow1 = mbase + 16 + l15;
  if (arow0 >= n_rows) arow0 = n_rows - 1;   // clamp loads; stores guarded
  if (arow1 >= n_rows) arow1 = n_rows - 1;
  const float* xrow0 = x + (size_t)arow0 * FIN;
  const float* xrow1 = x + (size_t)arow1 * FIN;

  v8f zero = {};
  v8f acc0[8], acc1[8];
#pragma unroll
  for (int t = 0; t < 8; ++t) { acc0[t] = zero; acc1[t] = zero; }

  const v16bf* wl = reinterpret_cast<const v16bf*>(lds_raw);

  for (int kk = 0; kk < FIN / 32; ++kk) {
    const int kb = kk * 32 + half * 8;    // A layout: K base 0 / 8 per lane half

    float4 q0 = *reinterpret_cast<const float4*>(xrow0 + kb);
    float4 q1 = *reinterpret_cast<const float4*>(xrow0 + kb + 4);
    float4 q2 = *reinterpret_cast<const float4*>(xrow0 + kb + 16);
    float4 q3 = *reinterpret_cast<const float4*>(xrow0 + kb + 20);
    float4 r0 = *reinterpret_cast<const float4*>(xrow1 + kb);
    float4 r1 = *reinterpret_cast<const float4*>(xrow1 + kb + 4);
    float4 r2 = *reinterpret_cast<const float4*>(xrow1 + kb + 16);
    float4 r3 = *reinterpret_cast<const float4*>(xrow1 + kb + 20);
    if (kk + 1 < FIN / 32) {
      __builtin_prefetch(xrow0 + kb + 32, 0, 0);   // global_prefetch_b8
      __builtin_prefetch(xrow1 + kb + 32, 0, 0);
    }

    v16bf a0, a1;
    a0[0]  = f2bf(q0.x); a0[1]  = f2bf(q0.y); a0[2]  = f2bf(q0.z); a0[3]  = f2bf(q0.w);
    a0[4]  = f2bf(q1.x); a0[5]  = f2bf(q1.y); a0[6]  = f2bf(q1.z); a0[7]  = f2bf(q1.w);
    a0[8]  = f2bf(q2.x); a0[9]  = f2bf(q2.y); a0[10] = f2bf(q2.z); a0[11] = f2bf(q2.w);
    a0[12] = f2bf(q3.x); a0[13] = f2bf(q3.y); a0[14] = f2bf(q3.z); a0[15] = f2bf(q3.w);
    a1[0]  = f2bf(r0.x); a1[1]  = f2bf(r0.y); a1[2]  = f2bf(r0.z); a1[3]  = f2bf(r0.w);
    a1[4]  = f2bf(r1.x); a1[5]  = f2bf(r1.y); a1[6]  = f2bf(r1.z); a1[7]  = f2bf(r1.w);
    a1[8]  = f2bf(r2.x); a1[9]  = f2bf(r2.y); a1[10] = f2bf(r2.z); a1[11] = f2bf(r2.w);
    a1[12] = f2bf(r3.x); a1[13] = f2bf(r3.y); a1[14] = f2bf(r3.z); a1[15] = f2bf(r3.w);

    // batch all 8 B fragments of this k-step into registers first
    v16bf bfr[8];
#pragma unroll
    for (int t = 0; t < 8; ++t)
      bfr[t] = wl[((size_t)(t * 16 + kk)) * 32 + lane];   // 2x ds_load_b128 each

    // uninterrupted WMMA burst: 16 matrix ops per k-step
#pragma unroll
    for (int t = 0; t < 8; ++t) {
      acc0[t] = __builtin_amdgcn_wmma_f32_16x16x32_bf16(
          false, a0, false, bfr[t], (short)0, acc0[t], false, false);
      acc1[t] = __builtin_amdgcn_wmma_f32_16x16x32_bf16(
          false, a1, false, bfr[t], (short)0, acc1[t], false, false);
    }
  }

  // C/D 16x16 f32 layout: VGPR r -> M = r + half*8, N = l15
#pragma unroll
  for (int t = 0; t < 8; ++t) {
    const int col = t * 16 + l15;
#pragma unroll
    for (int r = 0; r < 8; ++r) {
      const int row0 = mbase + r + half * 8;
      const int row1 = mbase + 16 + r + half * 8;
      if (row0 < n_rows) pre[(size_t)row0 * FOUT + col] = acc0[t][r];
      if (row1 < n_rows) pre[(size_t)row1 * FOUT + col] = acc1[t][r];
    }
  }
}

// ---------------------------------------------------------------------------
// SpMM scatter: one edge per wave32; lane covers 4 of 128 cols (float4 gather
// from L2-resident pre_sup, 4x global_atomic_add_f32 scatter).
// ---------------------------------------------------------------------------
__global__ void __launch_bounds__(256)
scatter_kernel(const long long* __restrict__ rows, const long long* __restrict__ cols,
               const float* __restrict__ vals, const float* __restrict__ pre,
               float* __restrict__ out, int nE) {
  int t    = blockIdx.x * blockDim.x + threadIdx.x;
  int e    = t >> 5;
  int lane = t & 31;
  if (e >= nE) return;
  int   r = (int)rows[e];
  int   c = (int)cols[e];
  float v = vals[e];
  const float4 m = *reinterpret_cast<const float4*>(pre + (size_t)c * FOUT + lane * 4);
  float* dst = out + (size_t)r * FOUT + lane * 4;
  __hip_atomic_fetch_add(dst + 0, v * m.x, __ATOMIC_RELAXED, __HIP_MEMORY_SCOPE_AGENT);
  __hip_atomic_fetch_add(dst + 1, v * m.y, __ATOMIC_RELAXED, __HIP_MEMORY_SCOPE_AGENT);
  __hip_atomic_fetch_add(dst + 2, v * m.z, __ATOMIC_RELAXED, __HIP_MEMORY_SCOPE_AGENT);
  __hip_atomic_fetch_add(dst + 3, v * m.w, __ATOMIC_RELAXED, __HIP_MEMORY_SCOPE_AGENT);
}

__global__ void __launch_bounds__(256)
zero_kernel(float* __restrict__ out, size_t n) {
  size_t i = (size_t)blockIdx.x * blockDim.x + threadIdx.x;
  if (i < n) out[i] = 0.0f;
}

__global__ void __launch_bounds__(256)
finalize_kernel(float* __restrict__ out, const float* __restrict__ bias, size_t n) {
  size_t i = (size_t)blockIdx.x * blockDim.x + threadIdx.x;
  if (i < n) {
    float v = out[i] + bias[i & (FOUT - 1)];
    out[i] = v > 0.0f ? v : 0.0f;
  }
}

// ---------------------------------------------------------------------------
extern "C" void kernel_launch(void* const* d_in, const int* in_sizes, int n_in,
                              void* d_out, int out_size, void* d_ws, size_t ws_size,
                              hipStream_t stream) {
  const float*     x     = (const float*)d_in[0];
  const long long* sidx  = (const long long*)d_in[1];   // int64 [2,E]
  const float*     svals = (const float*)d_in[2];
  const long long* oidx  = (const long long*)d_in[3];   // int64 [2,E]
  const float*     ovals = (const float*)d_in[4];
  const float*     W     = (const float*)d_in[5];
  const float*     bias  = (const float*)d_in[6];
  float*           out   = (float*)d_out;

  const int nE     = in_sizes[2];              // E
  const int n_rows = in_sizes[0] / FIN;        // N

  // workspace: [0, 128KB) bf16 W fragments ; [256KB, ...) pre_sup f32 (25.6MB)
  uint16_t* wfrag = (uint16_t*)d_ws;
  float*    pre   = (float*)((char*)d_ws + 256 * 1024);

  wprep_kernel<<<16, 256, 0, stream>>>(W, wfrag);

  const int mblocks = (n_rows + 255) / 256;
  gemm_kernel<<<mblocks, 256, WFRAG_BYTES, stream>>>(x, wfrag, pre, n_rows);

  const size_t total = (size_t)2 * n_rows * FOUT;
  zero_kernel<<<(unsigned)((total + 255) / 256), 256, 0, stream>>>(out, total);

  const int sblocks = (int)(((long long)nE * 32 + 255) / 256);
  scatter_kernel<<<sblocks, 256, 0, stream>>>(sidx, sidx + nE, svals, pre, out, nE);
  scatter_kernel<<<sblocks, 256, 0, stream>>>(oidx, oidx + nE, ovals, pre,
                                              out + (size_t)n_rows * FOUT, nE);

  finalize_kernel<<<(unsigned)((total + 255) / 256), 256, 0, stream>>>(out, bias, total);
}
